// TimeReversalFeatureExtractor_18794776887341
// MI455X (gfx1250) — compile-verified, matrix operand order checked by reading, not run on hardware
//
#include <hip/hip_runtime.h>
#include <math.h>

typedef _Float16 f16;
typedef __attribute__((ext_vector_type(16))) _Float16 v16h;
typedef __attribute__((ext_vector_type(8)))  float    v8f;

#define SRATE   16000
#define NFFT    512
#define HOP     128
#define WINL    512
#define NBANDS  8
#define NAZ     72
#define NEL     13
#define NDIR    (NAZ*NEL)       /* 936 */
#define F0BIN   10              /* first in-band rfft bin (300 Hz @ 31.25 Hz/bin) */
#define NF      103             /* bins 10..112 inclusive */
#define NFP     104             /* padded per real/imag part */
#define NPK     (2*NFP)         /* 208 cols of packed DFT kernel [kr | ki] */
#define NFRM    122             /* (16000-512)/128 + 1 */
#define NB      4
#define NC      8
#define NT_LEN  16000
#define CSOUND  343.0f
#define SHRINKF 0.85f
#define TEMPF   0.35f
#define EPSV    1e-8f
#define PI_F    3.14159265358979323846f
#define NDT     59              /* ceil(936/16) direction tiles */
#define NDT2    30              /* direction tile PAIRS per wave */
#define NMT     8               /* ceil(122/16) frame tiles */

// ---------------------------------------------------------------------------
// WMMA 16-bit operand register layout (ISA 7.12.2):
//   lane L holds A row m=L&15 / B col n=L&15; half h (0..15) of lane L maps to
//   K = (h&7) + 16*(h>=8) + 8*(L>=16).
// We store tiles in LDS already swizzled into this lane-major layout so each
// lane loads its whole v16h operand as 32 contiguous bytes (2x ds_load_b128).
//   element with reduction index k and row/col index mn lives at:
// ---------------------------------------------------------------------------
__device__ inline int swz(int k, int mn) {
    int lane = mn + (((k >> 3) & 1) << 4);
    int h    = (k & 7) + (((k >> 4) & 1) << 3);
    return (lane << 4) + h;
}

// ---------------------------------------------------------------------------
// Kernel 0: build f16 DFT kernel table [512 x 208] and per-(b,c,d) delays
// ---------------------------------------------------------------------------
__global__ void k_build_consts(f16* __restrict__ dftk, float* __restrict__ delays,
                               const float* __restrict__ mics) {
    int i = blockIdx.x * blockDim.x + threadIdx.x;
    if (i < WINL * NPK) {
        int w = i / NPK, n = i % NPK;
        float win = 0.5f - 0.5f * __cosf(2.f * PI_F * (float)w / (float)WINL);
        int part = n / NFP;                 // 0 = real (kr), 1 = imag (ki)
        int f = n % NFP;
        float val = 0.f;
        if (f < NF) {
            float ph = 2.f * PI_F * (float)(F0BIN + f) * (float)w / (float)NFFT;
            val = (part == 0 ? __cosf(ph) : -__sinf(ph)) * win;
        }
        dftk[i] = (f16)val;
    }
    if (i < NB * NC * NDIR) {
        int d = i % NDIR; int bc = i / NDIR;
        int c = bc % NC;  int b  = bc / NC;
        float mx = 0.f, my = 0.f, mz = 0.f;
        for (int cc = 0; cc < NC; ++cc) {
            mx += mics[(b*NC+cc)*3+0];
            my += mics[(b*NC+cc)*3+1];
            mz += mics[(b*NC+cc)*3+2];
        }
        mx *= (1.f/NC); my *= (1.f/NC); mz *= (1.f/NC);
        float px = mics[(b*NC+c)*3+0] - mx;
        float py = mics[(b*NC+c)*3+1] - my;
        float pz = mics[(b*NC+c)*3+2] - mz;
        int el = d / NAZ, az = d % NAZ;
        float elr = (30.f + 10.f*(float)el) * (PI_F/180.f);
        float azr = (5.f  * (float)az)      * (PI_F/180.f);
        float se = __sinf(elr), ce = __cosf(elr);
        float dx = se * __cosf(azr), dy = se * __sinf(azr), dz = ce;
        delays[i] = -(px*dx + py*dy + pz*dz) * (1.f/CSOUND);
    }
}

// ---------------------------------------------------------------------------
// Kernel 1: conv-STFT via WMMA.  One wave per (b,c, M-tile, N-tile).
//   A = frames[t, w] (f16), B = dftk[w, 2F packed], K = 512 in 16 steps.
//   Tiles staged in LDS pre-swizzled to the WMMA register layout.
// ---------------------------------------------------------------------------
__global__ __launch_bounds__(32)
void k_stft_wmma(const float* __restrict__ audio, const f16* __restrict__ dftk,
                 float* __restrict__ res_r, float* __restrict__ res_i) {
    __shared__ alignas(32) f16 As[512];
    __shared__ alignas(32) f16 Bs[512];
    int id = blockIdx.x;
    int nt = id % 13; id /= 13;
    int mt = id % NMT; id /= NMT;
    int c  = id % NC; int b = id / NC;
    int lane = threadIdx.x;
    const size_t abase = (size_t)(b*NC + c) * NT_LEN;
    v8f acc = {};
    for (int ks = 0; ks < 16; ++ks) {
        // prefetch next K-step of this wave's audio rows into cache
        if (ks + 1 < 16) {
            int tp = mt*16 + (lane & 15);
            if (tp < NFRM)
                __builtin_prefetch(&audio[abase + tp*HOP + (ks+1)*32], 0, 3);
        }
        for (int j = lane; j < 512; j += 32) {
            int m = j >> 5, k = j & 31;
            int t = mt*16 + m;
            float v = 0.f;
            if (t < NFRM) v = audio[abase + t*HOP + ks*32 + k];
            As[swz(k, m)] = (f16)v;
        }
        for (int j = lane; j < 512; j += 32) {
            int k = j >> 4, n = j & 15;
            Bs[swz(k, n)] = dftk[(ks*32 + k)*NPK + (nt*16 + n)];
        }
        __syncthreads();
        v16h av = *reinterpret_cast<const v16h*>(&As[lane << 4]);
        v16h bv = *reinterpret_cast<const v16h*>(&Bs[lane << 4]);
        acc = __builtin_amdgcn_wmma_f32_16x16x32_f16(false, av, false, bv,
                                                     (short)0, acc, false, false);
        __syncthreads();
    }
    // D layout: lane L -> col (L&15); VGPR v -> row v + (L>=16 ? 8 : 0)
    int ncol = nt*16 + (lane & 15);
    int part = ncol / NFP; int f = ncol % NFP;
    if (f < NF) {
        float* dst = (part == 0) ? res_r : res_i;
        #pragma unroll
        for (int v = 0; v < 8; ++v) {
            int t = mt*16 + v + ((lane >= 16) ? 8 : 0);
            if (t < NFRM)
                dst[((size_t)(b*NC+c)*NF + f)*NFRM + t] = acc[v];
        }
    }
}

// ---------------------------------------------------------------------------
// Kernel 2: conj + PHAT whitening, in place
// ---------------------------------------------------------------------------
__global__ void k_phat(float* __restrict__ rr, float* __restrict__ ri, int n) {
    int i = blockIdx.x * blockDim.x + threadIdx.x;
    if (i >= n) return;
    float a = rr[i], bq = ri[i];
    float mag = fmaxf(sqrtf(a*a + bq*bq), EPSV);
    rr[i] =  a / mag;
    ri[i] = -bq / mag;   // conj then whiten
}

__global__ void k_zero(float* __restrict__ p, int n) {
    int i = blockIdx.x * blockDim.x + threadIdx.x;
    if (i < n) p[i] = 0.f;
}

// ---------------------------------------------------------------------------
// Kernel 3: beamforming power via WMMA.  One wave per (b, f, M-tile, D-tile
// PAIR) -- the A (residual) tile is staged once and reused for 4 WMMAs.
//   Complex GEMM over C=8 expressed as real K=16 (padded to 32):
//   A row t: [res_r(c=0..7) | res_i(c=0..7) | 0...].
//   Br col d: [cos | -sin | 0], Bi col d: [sin | cos | 0] via on-the-fly sincos.
//   power = beam_r^2 + beam_i^2, summed over t, atomic-added to fmap[b,f,d].
// ---------------------------------------------------------------------------
__global__ __launch_bounds__(32)
void k_beam_wmma(const float* __restrict__ res_r, const float* __restrict__ res_i,
                 const float* __restrict__ delays, float* __restrict__ fmap) {
    __shared__ alignas(32) f16 As[512];
    __shared__ alignas(32) f16 Br0[512];
    __shared__ alignas(32) f16 Bi0[512];
    __shared__ alignas(32) f16 Br1[512];
    __shared__ alignas(32) f16 Bi1[512];
    __shared__ float pwr0[32];
    __shared__ float pwr1[32];
    int id = blockIdx.x;
    int dp = id % NDT2; id /= NDT2;
    int mt = id % NMT; id /= NMT;
    int f  = id % NF;  int b = id / NF;
    int lane = threadIdx.x;
    float freq = (float)(F0BIN + f) * ((float)SRATE / (float)NFFT);

    // stage A (swizzled) + zero-fill B tiles
    for (int j = lane; j < 512; j += 32) {
        int m = j >> 5, k = j & 31;
        int t = mt*16 + m;
        float v = 0.f;
        if (t < NFRM && k < 16) {
            const float* src = (k < 8) ? res_r : res_i;
            v = src[((size_t)(b*NC + (k & 7))*NF + f)*NFRM + t];
        }
        As[swz(k, m)] = (f16)v;
        Br0[j] = (f16)0.f; Bi0[j] = (f16)0.f;
        Br1[j] = (f16)0.f; Bi1[j] = (f16)0.f;
    }
    __syncthreads();
    // steering for the two direction tiles (rows k<16; k>=16 stay zero)
    for (int j = lane; j < 8*16; j += 32) {
        int c = j >> 4, n = j & 15;
        int d0 = (dp*2)  *16 + n;
        int d1 = (dp*2+1)*16 + n;
        if (d0 < NDIR) {
            float ph = 2.f * PI_F * freq * delays[(b*NC + c)*NDIR + d0];
            float s = __sinf(ph), cs = __cosf(ph);
            Br0[swz(c,   n)] = (f16)cs;
            Br0[swz(c+8, n)] = (f16)(-s);
            Bi0[swz(c,   n)] = (f16)s;
            Bi0[swz(c+8, n)] = (f16)cs;
        }
        if (d1 < NDIR) {
            float ph = 2.f * PI_F * freq * delays[(b*NC + c)*NDIR + d1];
            float s = __sinf(ph), cs = __cosf(ph);
            Br1[swz(c,   n)] = (f16)cs;
            Br1[swz(c+8, n)] = (f16)(-s);
            Bi1[swz(c,   n)] = (f16)s;
            Bi1[swz(c+8, n)] = (f16)cs;
        }
    }
    __syncthreads();

    v16h av  = *reinterpret_cast<const v16h*>(&As [lane << 4]);
    v16h br0 = *reinterpret_cast<const v16h*>(&Br0[lane << 4]);
    v16h bi0 = *reinterpret_cast<const v16h*>(&Bi0[lane << 4]);
    v16h br1 = *reinterpret_cast<const v16h*>(&Br1[lane << 4]);
    v16h bi1 = *reinterpret_cast<const v16h*>(&Bi1[lane << 4]);
    v8f cr0 = {}, ci0 = {}, cr1 = {}, ci1 = {};
    cr0 = __builtin_amdgcn_wmma_f32_16x16x32_f16(false, av, false, br0, (short)0, cr0, false, false);
    ci0 = __builtin_amdgcn_wmma_f32_16x16x32_f16(false, av, false, bi0, (short)0, ci0, false, false);
    cr1 = __builtin_amdgcn_wmma_f32_16x16x32_f16(false, av, false, br1, (short)0, cr1, false, false);
    ci1 = __builtin_amdgcn_wmma_f32_16x16x32_f16(false, av, false, bi1, (short)0, ci1, false, false);

    float p0 = 0.f, p1 = 0.f;
    #pragma unroll
    for (int v = 0; v < 8; ++v) {
        int t = mt*16 + v + ((lane >= 16) ? 8 : 0);
        if (t < NFRM) {
            p0 += cr0[v]*cr0[v] + ci0[v]*ci0[v];
            p1 += cr1[v]*cr1[v] + ci1[v]*ci1[v];
        }
    }
    pwr0[lane] = p0;
    pwr1[lane] = p1;
    __syncthreads();
    if (lane < 16) {
        int d0 = (dp*2)  *16 + lane;
        int d1 = (dp*2+1)*16 + lane;
        float* row = &fmap[((size_t)b*NF + f)*NDIR];
        if (d0 < NDIR) atomicAdd(&row[d0], pwr0[lane] + pwr0[lane + 16]);
        if (d1 < NDIR) atomicAdd(&row[d1], pwr1[lane] + pwr1[lane + 16]);
    }
}

// ---------------------------------------------------------------------------
// Small f32 reduction / update kernels
// ---------------------------------------------------------------------------
__global__ void k_pmean(const float* __restrict__ fmap, float* __restrict__ pmean) {
    int i = blockIdx.x * blockDim.x + threadIdx.x;
    if (i >= NB * NDIR) return;
    int d = i % NDIR, b = i / NDIR;
    float s = 0.f;
    for (int f = 0; f < NF; ++f) s += fmap[((size_t)b*NF + f)*NDIR + d];
    pmean[i] = s / (float)(NF * NFRM);
}

__global__ __launch_bounds__(256)
void k_softmax(const float* __restrict__ pmean, float* __restrict__ wsm) {
    __shared__ float red[256];
    int b = blockIdx.x, tid = threadIdx.x;
    float mx = -1e30f;
    for (int d = tid; d < NDIR; d += 256) mx = fmaxf(mx, pmean[b*NDIR + d] / TEMPF);
    red[tid] = mx; __syncthreads();
    for (int s = 128; s > 0; s >>= 1) { if (tid < s) red[tid] = fmaxf(red[tid], red[tid+s]); __syncthreads(); }
    mx = red[0]; __syncthreads();
    float sum = 0.f;
    for (int d = tid; d < NDIR; d += 256) sum += __expf(pmean[b*NDIR + d] / TEMPF - mx);
    red[tid] = sum; __syncthreads();
    for (int s = 128; s > 0; s >>= 1) { if (tid < s) red[tid] += red[tid+s]; __syncthreads(); }
    float tot = red[0];
    for (int d = tid; d < NDIR; d += 256)
        wsm[b*NDIR + d] = __expf(pmean[b*NDIR + d] / TEMPF - mx) / tot;
}

__global__ void k_bandmap(const float* __restrict__ fmap, float* __restrict__ bandmap) {
    int i = blockIdx.x * blockDim.x + threadIdx.x;
    if (i >= NB * NBANDS * NDIR) return;
    int d = i % NDIR; int bg = i / NDIR;
    int g = bg % NBANDS; int b = bg / NBANDS;
    float s = 0.f; int cnt = 0;
    for (int f = 0; f < NF; ++f) {
        if ((f * NBANDS) / NF == g) { s += fmap[((size_t)b*NF + f)*NDIR + d]; ++cnt; }
    }
    float m = s / ((float)cnt * (float)NFRM);
    bandmap[i] = log1pf(m);
}

__global__ __launch_bounds__(256)
void k_bmax(const float* __restrict__ bandmap, float* __restrict__ maxv) {
    __shared__ float red[256];
    int b = blockIdx.x, tid = threadIdx.x;
    float mx = 0.f;
    for (int j = tid; j < NBANDS * NDIR; j += 256)
        mx = fmaxf(mx, bandmap[(size_t)b*NBANDS*NDIR + j]);
    red[tid] = mx; __syncthreads();
    for (int s = 128; s > 0; s >>= 1) { if (tid < s) red[tid] = fmaxf(red[tid], red[tid+s]); __syncthreads(); }
    if (tid == 0) maxv[b] = red[0];
}

__global__ void k_writeout(const float* __restrict__ bandmap, const float* __restrict__ maxv,
                           float* __restrict__ out) {
    int i = blockIdx.x * blockDim.x + threadIdx.x;
    if (i >= NB * NBANDS * NEL * NAZ) return;
    int az = i % NAZ; int r = i / NAZ;
    int el = r % NEL; r /= NEL;
    int g  = r % NBANDS; int b = r / NBANDS;
    float m = bandmap[((size_t)(b*NBANDS) + g)*NDIR + el*NAZ + az];
    out[i] = m / fmaxf(maxv[b], EPSV);
}

// a[b,c,f] = sum_d w[b,d] * exp(i*2*pi*freq_f*delay[b,c,d])
__global__ void k_acomp(const float* __restrict__ wsm, const float* __restrict__ delays,
                        float* __restrict__ a_r, float* __restrict__ a_i) {
    int i = blockIdx.x * blockDim.x + threadIdx.x;
    if (i >= NB * NC * NF) return;
    int f = i % NF; int bc = i / NF;
    int b = bc / NC;
    float freq = (float)(F0BIN + f) * ((float)SRATE / (float)NFFT);
    float ar = 0.f, ai = 0.f;
    for (int d = 0; d < NDIR; ++d) {
        float w = wsm[b*NDIR + d];
        float ph = 2.f * PI_F * freq * delays[bc*NDIR + d];
        ar += w * __cosf(ph);
        ai += w * __sinf(ph);
    }
    a_r[i] = ar; a_i[i] = ai;
}

__global__ void k_normc(const float* __restrict__ a_r, const float* __restrict__ a_i,
                        float* __restrict__ normf) {
    int i = blockIdx.x * blockDim.x + threadIdx.x;
    if (i >= NB * NF) return;
    int f = i % NF, b = i / NF;
    float s = 0.f;
    for (int c = 0; c < NC; ++c) {
        float ar = a_r[(b*NC + c)*NF + f], ai = a_i[(b*NC + c)*NF + f];
        s += ar*ar + ai*ai;
    }
    normf[i] = s;
}

// res -= SHRINK * a * s_hat, s_hat = sum_c conj(a)*res / max(norm, eps)
__global__ void k_update(const float* __restrict__ a_r, const float* __restrict__ a_i,
                         const float* __restrict__ normf,
                         float* __restrict__ res_r, float* __restrict__ res_i) {
    int i = blockIdx.x * blockDim.x + threadIdx.x;
    if (i >= NB * NF * NFRM) return;
    int t = i % NFRM; int bf = i / NFRM;
    int f = bf % NF, b = bf / NF;
    float sr = 0.f, si = 0.f;
    for (int c = 0; c < NC; ++c) {
        float ar = a_r[(b*NC + c)*NF + f], ai = a_i[(b*NC + c)*NF + f];
        size_t idx = ((size_t)(b*NC + c)*NF + f)*NFRM + t;
        float rr = res_r[idx], ri = res_i[idx];
        sr += ar*rr + ai*ri;        // conj(a)*res, real
        si += ar*ri - ai*rr;        // conj(a)*res, imag
    }
    float nrm = fmaxf(normf[b*NF + f], EPSV);
    sr /= nrm; si /= nrm;
    for (int c = 0; c < NC; ++c) {
        float ar = a_r[(b*NC + c)*NF + f], ai = a_i[(b*NC + c)*NF + f];
        size_t idx = ((size_t)(b*NC + c)*NF + f)*NFRM + t;
        res_r[idx] -= SHRINKF * (ar*sr - ai*si);
        res_i[idx] -= SHRINKF * (ar*si + ai*sr);
    }
}

// ---------------------------------------------------------------------------
extern "C" void kernel_launch(void* const* d_in, const int* in_sizes, int n_in,
                              void* d_out, int out_size, void* d_ws, size_t ws_size,
                              hipStream_t stream) {
    (void)in_sizes; (void)n_in; (void)out_size; (void)ws_size;
    const float* audio = (const float*)d_in[0];
    const float* mics  = (const float*)d_in[1];
    float* out = (float*)d_out;

    char* p = (char*)d_ws;
    auto alloc = [&](size_t bytes) -> void* {
        void* r = (void*)p;
        p += (bytes + 255) & ~(size_t)255;
        return r;
    };
    f16*   dftk    = (f16*)  alloc((size_t)WINL*NPK*sizeof(f16));
    float* delays  = (float*)alloc((size_t)NB*NC*NDIR*sizeof(float));
    float* res_r   = (float*)alloc((size_t)NB*NC*NF*NFRM*sizeof(float));
    float* res_i   = (float*)alloc((size_t)NB*NC*NF*NFRM*sizeof(float));
    float* fmap    = (float*)alloc((size_t)NB*NF*NDIR*sizeof(float));
    float* pmean   = (float*)alloc((size_t)NB*NDIR*sizeof(float));
    float* wsm     = (float*)alloc((size_t)NB*NDIR*sizeof(float));
    float* bandmap = (float*)alloc((size_t)NB*NBANDS*NDIR*sizeof(float));
    float* maxv    = (float*)alloc((size_t)NB*sizeof(float));
    float* a_r     = (float*)alloc((size_t)NB*NC*NF*sizeof(float));
    float* a_i     = (float*)alloc((size_t)NB*NC*NF*sizeof(float));
    float* normf   = (float*)alloc((size_t)NB*NF*sizeof(float));

    k_build_consts<<<(WINL*NPK + 255)/256, 256, 0, stream>>>(dftk, delays, mics);
    k_stft_wmma<<<NB*NC*NMT*13, 32, 0, stream>>>(audio, dftk, res_r, res_i);
    int nspec = NB*NC*NF*NFRM;
    k_phat<<<(nspec + 255)/256, 256, 0, stream>>>(res_r, res_i, nspec);

    for (int it = 0; it < 3; ++it) {
        int nfm = NB*NF*NDIR;
        k_zero<<<(nfm + 255)/256, 256, 0, stream>>>(fmap, nfm);
        k_beam_wmma<<<NB*NF*NMT*NDT2, 32, 0, stream>>>(res_r, res_i, delays, fmap);
        // map output (written every iteration; last iteration's values survive)
        k_bandmap<<<(NB*NBANDS*NDIR + 255)/256, 256, 0, stream>>>(fmap, bandmap);
        k_bmax<<<NB, 256, 0, stream>>>(bandmap, maxv);
        k_writeout<<<(NB*NBANDS*NEL*NAZ + 255)/256, 256, 0, stream>>>(bandmap, maxv, out);
        // dominant-direction shrink-subtract
        k_pmean<<<(NB*NDIR + 255)/256, 256, 0, stream>>>(fmap, pmean);
        k_softmax<<<NB, 256, 0, stream>>>(pmean, wsm);
        k_acomp<<<(NB*NC*NF + 255)/256, 256, 0, stream>>>(wsm, delays, a_r, a_i);
        k_normc<<<(NB*NF + 255)/256, 256, 0, stream>>>(a_r, a_i, normf);
        k_update<<<(NB*NF*NFRM + 255)/256, 256, 0, stream>>>(a_r, a_i, normf, res_r, res_i);
    }
}
